// GCN_28716151341438
// MI455X (gfx1250) — compile-verified, matrix operand order checked by reading, not run on hardware
//
#include <hip/hip_runtime.h>
#include <hip/hip_bf16.h>
#include <stdint.h>

#define Nn_ 16384
#define E_  262144
#define H_  512
#define G_  256
#define L_  4
#define OUT_ 128
#define AF_ 9
#define AV_ 128
#define BF_ 3
#define BV_ 8

typedef __attribute__((ext_vector_type(16))) __bf16 v16bf;
typedef __attribute__((ext_vector_type(8)))  float  v8f;

__device__ __forceinline__ unsigned short f32_to_bf16(float f) {
  unsigned int u = __float_as_uint(f);
  unsigned int r = (u + 0x7FFFu + ((u >> 16) & 1u)) >> 16;   // round-nearest-even
  return (unsigned short)r;
}

union FragB {
  v16bf v;
  unsigned int u[8];
};

// CDNA5 async global->LDS copy, 16B per lane, tracked by ASYNCcnt (ISA §08).
__device__ __forceinline__ void async_copy16(unsigned lds_off, unsigned long long gaddr) {
  asm volatile("global_load_async_to_lds_b128 %0, %1, off"
               :: "v"(lds_off), "v"(gaddr) : "memory");
}
__device__ __forceinline__ void wait_async_le4() {
  asm volatile("s_wait_asynccnt 4" ::: "memory");
}
__device__ __forceinline__ void wait_async_0() {
  asm volatile("s_wait_asynccnt 0" ::: "memory");
}

// ---------------------------------------------------------------- utilities
__global__ void zero_f32(float* __restrict__ p, int n) {
  int i = blockIdx.x * blockDim.x + threadIdx.x;
  if (i < n) p[i] = 0.0f;
}

__global__ void zero_i32(int* __restrict__ p, int n) {
  int i = blockIdx.x * blockDim.x + threadIdx.x;
  if (i < n) p[i] = 0;
}

__global__ void deg_count(const int* __restrict__ dst, int* __restrict__ deg) {
  int e = blockIdx.x * blockDim.x + threadIdx.x;
  if (e < E_) atomicAdd(&deg[dst[e]], 1);
}

__global__ void deg_recip(const int* __restrict__ deg, float* __restrict__ inv) {
  int n = blockIdx.x * blockDim.x + threadIdx.x;
  if (n < Nn_) inv[n] = 1.0f / fmaxf((float)deg[n], 1.0f);
}

__global__ void bounds_init(int* __restrict__ gs, int* __restrict__ ge) {
  int g = threadIdx.x;
  if (g < G_) { gs[g] = Nn_; ge[g] = 0; }
}

__global__ void bounds_scan(const int* __restrict__ gid, int* __restrict__ gs,
                            int* __restrict__ ge) {
  int n = blockIdx.x * blockDim.x + threadIdx.x;
  if (n < Nn_) {
    int g = gid[n];
    atomicMin(&gs[g], n);
    atomicMax(&ge[g], n + 1);
  }
}

// Wt[n*K + k] = bf16(Win[k*N + n])  (pre-transpose+convert weights, one shot)
__global__ void transpose_bf16(const float* __restrict__ Win,
                               unsigned short* __restrict__ Wt, int K, int N) {
  int idx = blockIdx.x * blockDim.x + threadIdx.x;
  if (idx >= K * N) return;
  int n = idx / K, k = idx % K;         // k contiguous -> coalesced writes
  Wt[(size_t)n * K + k] = f32_to_bf16(Win[(size_t)k * N + n]);
}

// ------------------------------------------------------------- atom encoder
__global__ void atom_encode(const int* __restrict__ hn,
                            const float* __restrict__ aemb,
                            float* __restrict__ h) {
  int tid = blockIdx.x * blockDim.x + threadIdx.x;  // Nn_ * 128 threads
  int n = tid >> 7;
  int c = (tid & 127) << 2;
  if (n >= Nn_) return;
  float s0 = 0.f, s1 = 0.f, s2 = 0.f, s3 = 0.f;
#pragma unroll
  for (int f = 0; f < AF_; ++f) {
    int idx = hn[n * AF_ + f];
    const float* p = &aemb[((size_t)f * AV_ + idx) * H_ + c];
    s0 += p[0]; s1 += p[1]; s2 += p[2]; s3 += p[3];
  }
  float* o = &h[(size_t)n * H_ + c];
  o[0] = s0; o[1] = s1; o[2] = s2; o[3] = s3;
}

// ------------------------------------------ bond encode + gather + scatter
__global__ void edge_msg(const float* __restrict__ h,
                         const float* __restrict__ bemb,   // [BF_][BV_][H_] layer i
                         const int* __restrict__ h_edge,
                         const int* __restrict__ src,
                         const int* __restrict__ dst,
                         float* __restrict__ neigh) {
  int tid = blockIdx.x * blockDim.x + threadIdx.x;  // E_ * 128 threads
  int e = tid >> 7;
  int c = (tid & 127) << 2;
  if (e >= E_) return;
  int i0 = h_edge[e * BF_ + 0];
  int i1 = h_edge[e * BF_ + 1];
  int i2 = h_edge[e * BF_ + 2];
  const float* b0 = &bemb[((size_t)0 * BV_ + i0) * H_ + c];
  const float* b1 = &bemb[((size_t)1 * BV_ + i1) * H_ + c];
  const float* b2 = &bemb[((size_t)2 * BV_ + i2) * H_ + c];
  const float* hs = &h[(size_t)src[e] * H_ + c];
  float* o = &neigh[(size_t)dst[e] * H_ + c];
#pragma unroll
  for (int k = 0; k < 4; ++k) {
    float m = hs[k] + b0[k] + b1[k] + b2[k];
    atomicAdd(&o[k], m);
  }
}

// ------------------------- rin = h + neigh * invdeg, emitted directly as bf16
__global__ void combine_bf16(const float* __restrict__ h, const float* __restrict__ inv,
                             const float* __restrict__ neigh,
                             unsigned short* __restrict__ rbf) {
  int tid = blockIdx.x * blockDim.x + threadIdx.x;
  int n = tid >> 7;
  int c = (tid & 127) << 2;
  if (n >= Nn_) return;
  float iv = inv[n];
  const float* hp = &h[(size_t)n * H_ + c];
  const float* np = &neigh[(size_t)n * H_ + c];
  unsigned short* rp = &rbf[(size_t)n * H_ + c];
#pragma unroll
  for (int k = 0; k < 4; ++k) rp[k] = f32_to_bf16(hp[k] + np[k] * iv);
}

// --------------------------------------------------------- WMMA bf16 GEMM
// C[M,N] = A[M,K] * Bt[N,K]^T + bias[N]; bf16 inputs, f32 accum/output.
// 256 threads = 8 waves; block tile 128x128x32; double-buffered LDS filled by
// GLOBAL_LOAD_ASYNC_TO_LDS_B128 (ASYNCcnt software pipeline). Fragment loads
// are fenced from the MAC chain so the 8 WMMAs issue back-to-back.
#define BM 128
#define BN 128
#define BK 32
__global__ __launch_bounds__(256) void wmma_gemm_bf16(
    const unsigned short* __restrict__ A,   // [M,K] bf16
    const unsigned short* __restrict__ Bt,  // [N,K] bf16 (pre-transposed)
    const float* __restrict__ bias, float* __restrict__ C,
    int M, int N, int K) {
  __shared__ unsigned short As[2][BM][BK];
  __shared__ unsigned short Bs[2][BN][BK];

  int tid  = threadIdx.x;
  int wave = tid >> 5;
  int lane = tid & 31;
  int n0 = blockIdx.x * BN;
  int m0 = blockIdx.y * BM;

  v8f acc[8];
#pragma unroll
  for (int j = 0; j < 8; ++j)
#pragma unroll
    for (int r = 0; r < 8; ++r) acc[j][r] = 0.0f;

  // LDS byte offsets: low 32 bits of the generic address are the LDS address.
  unsigned asBase = (unsigned)(uintptr_t)&As[0][0][0];
  unsigned bsBase = (unsigned)(uintptr_t)&Bs[0][0][0];
  const unsigned stageBytes = BM * BK * 2;   // 8KB per stage per matrix

  // Copy plan: tile is 512 x 16B chunks; 256 threads -> 2 chunks each / matrix.
  int r0 = tid >> 2,         p0 = tid & 3;
  int r1 = (tid + 256) >> 2, p1 = tid & 3;   // (tid+256)&3 == tid&3

  unsigned long long aBase = (unsigned long long)(uintptr_t)A;
  unsigned long long bBase = (unsigned long long)(uintptr_t)Bt;

#define ISSUE_STAGE(buf, k0)                                                     \
  do {                                                                           \
    async_copy16(asBase + (buf) * stageBytes + (r0 * BK + p0 * 8) * 2,           \
                 aBase + ((size_t)(m0 + r0) * K + (k0) + p0 * 8) * 2);           \
    async_copy16(bsBase + (buf) * stageBytes + (r0 * BK + p0 * 8) * 2,           \
                 bBase + ((size_t)(n0 + r0) * K + (k0) + p0 * 8) * 2);           \
    async_copy16(asBase + (buf) * stageBytes + (r1 * BK + p1 * 8) * 2,           \
                 aBase + ((size_t)(m0 + r1) * K + (k0) + p1 * 8) * 2);           \
    async_copy16(bsBase + (buf) * stageBytes + (r1 * BK + p1 * 8) * 2,           \
                 bBase + ((size_t)(n0 + r1) * K + (k0) + p1 * 8) * 2);           \
  } while (0)

  int lm  = lane & 15;
  int kh  = lane >> 4;        // which K half-group this lane owns
  int hi8 = kh * 8;

  // Compute one staged 128x128x32 step from LDS buffer `buf`.
  auto compute_stage = [&](int buf) {
    // A fragment (ISA 16-bit A layout): lane(m=lm) holds K[kh*8..+8) in v0..3
    // and K[16+kh*8..+8) in v4..7; two 16B-aligned ds_load_b128 each.
    FragB a;
    const unsigned short* ap = &As[buf][wave * 16 + lm][0];
    *(uint4*)&a.u[0] = *(const uint4*)&ap[kh * 8];
    *(uint4*)&a.u[4] = *(const uint4*)&ap[16 + kh * 8];

    FragB bfr[8];
#pragma unroll
    for (int j = 0; j < 8; ++j) {
      const unsigned short* bp = &Bs[buf][j * 16 + lm][0];
      *(uint4*)&bfr[j].u[0] = *(const uint4*)&bp[kh * 8];
      *(uint4*)&bfr[j].u[4] = *(const uint4*)&bp[16 + kh * 8];
    }
    // Keep all 18 ds_load_b128 ahead of the MAC chain -> one s_wait_dscnt,
    // then 8 back-to-back v_wmma (no per-MAC LDS round-trips).
    __builtin_amdgcn_sched_barrier(0);
#pragma unroll
    for (int j = 0; j < 8; ++j) {
      acc[j] = __builtin_amdgcn_wmma_f32_16x16x32_bf16(
          false, a.v, false, bfr[j].v, (short)0, acc[j], false, false);
    }
  };

  int steps = K / BK;
  ISSUE_STAGE(0, 0);

  // Steady state: branch-free body; last step peeled.
  for (int s = 0; s < steps - 1; ++s) {
    ISSUE_STAGE((s + 1) & 1, (s + 1) * BK);
    wait_async_le4();          // oldest 4 (stage s) retired; per-wave FIFO order
    __syncthreads();
    compute_stage(s & 1);
    __syncthreads();
  }
  wait_async_0();
  __syncthreads();
  compute_stage((steps - 1) & 1);
#undef ISSUE_STAGE

  // Epilogue: C layout — VGPR r: lanes 0-15 row M=r, lanes 16-31 row M=8+r.
#pragma unroll
  for (int j = 0; j < 8; ++j) {
    int col = n0 + j * 16 + lm;
    float bv = bias ? bias[col] : 0.0f;
#pragma unroll
    for (int r = 0; r < 8; ++r) {
      int row = m0 + wave * 16 + hi8 + r;
      C[(size_t)row * N + col] = acc[j][r] + bv;
    }
  }
}

// ------------------------------------------------------- GraphNorm stats
// One workgroup per graph; nodes of a graph are contiguous (gid sorted).
// var(h - a*mean) = E[h^2] - 2a*mean^2 + a^2*mean^2 (single pass).
__global__ __launch_bounds__(256) void gn_stats(
    const float* __restrict__ r, const int* __restrict__ gs,
    const int* __restrict__ ge, const float* __restrict__ ms,
    float* __restrict__ mean, float* __restrict__ var) {
  int g = blockIdx.x;
  int s = gs[g], e = ge[g];
  float cnt = fmaxf((float)(e - s), 1.0f);
  for (int f = threadIdx.x; f < H_; f += 256) {
    float sum = 0.f, ss = 0.f;
    for (int n = s; n < e; ++n) {
      float v = r[(size_t)n * H_ + f];
      sum += v; ss += v * v;
    }
    float m = sum / cnt;
    float a = ms[f];
    mean[g * H_ + f] = m;
    var[g * H_ + f] = ss / cnt - 2.f * a * m * m + a * a * m * m;
  }
}

__global__ void gn_apply(const float* __restrict__ r, const float* __restrict__ x,
                         const int* __restrict__ gid,
                         const float* __restrict__ mean, const float* __restrict__ var,
                         const float* __restrict__ gw, const float* __restrict__ gb,
                         const float* __restrict__ ms, float* __restrict__ hout,
                         int relu) {
  int tid = blockIdx.x * blockDim.x + threadIdx.x;
  int n = tid >> 7;
  int c = (tid & 127) << 2;
  if (n >= Nn_) return;
  int g = gid[n];
#pragma unroll
  for (int k = 0; k < 4; ++k) {
    int f = c + k;
    float m = mean[g * H_ + f];
    float v = var[g * H_ + f];
    float o = gw[f] * (r[(size_t)n * H_ + f] - m * ms[f]) * rsqrtf(v + 1e-6f) + gb[f];
    if (relu) o = fmaxf(o, 0.0f);
    hout[(size_t)n * H_ + f] = o + x[(size_t)n * H_ + f];
  }
}

// ---------------------------------------------- pooling (emits bf16 for GEMM)
__global__ __launch_bounds__(256) void pool_sum_bf16(const float* __restrict__ h,
                                                     const int* __restrict__ gs,
                                                     const int* __restrict__ ge,
                                                     unsigned short* __restrict__ pooled) {
  int g = blockIdx.x;
  int s = gs[g], e = ge[g];
  for (int f = threadIdx.x; f < H_; f += 256) {
    float sum = 0.f;
    for (int n = s; n < e; ++n) sum += h[(size_t)n * H_ + f];
    pooled[g * H_ + f] = f32_to_bf16(sum);
  }
}

// ================================================================ launcher
extern "C" void kernel_launch(void* const* d_in, const int* in_sizes, int n_in,
                              void* d_out, int out_size, void* d_ws, size_t ws_size,
                              hipStream_t stream) {
  const int*   h_node   = (const int*)d_in[0];
  const int*   h_edge   = (const int*)d_in[1];
  const int*   src      = (const int*)d_in[2];
  const int*   dst      = (const int*)d_in[3];
  const int*   gid      = (const int*)d_in[4];
  const float* atom_emb = (const float*)d_in[5];
  const float* bond_emb = (const float*)d_in[6];
  const float* W        = (const float*)d_in[7];
  const float* bb       = (const float*)d_in[8];
  const float* gn_w     = (const float*)d_in[9];
  const float* gn_b     = (const float*)d_in[10];
  const float* gn_ms    = (const float*)d_in[11];
  const float* Wp       = (const float*)d_in[12];
  const float* bp       = (const float*)d_in[13];
  float* out = (float*)d_out;

  const size_t NH = (size_t)Nn_ * H_;
  float* h0     = (float*)d_ws;
  float* h1     = h0 + NH;
  float* neigh  = h1 + NH;
  float* rbuf   = neigh + NH;                    // GEMM f32 output
  int*   deg    = (int*)(rbuf + NH);
  float* invdeg = (float*)(deg + Nn_);
  int*   gs     = (int*)(invdeg + Nn_);
  int*   ge     = gs + G_;
  float* mean   = (float*)(ge + G_);
  float* var    = mean + (size_t)G_ * H_;
  unsigned short* rbf    = (unsigned short*)(var + (size_t)G_ * H_);  // [N,H] bf16
  unsigned short* wt     = rbf + NH;                    // L_ x [H,H] bf16 (transposed)
  unsigned short* wpt    = wt + (size_t)L_ * H_ * H_;   // [OUT,H] bf16
  unsigned short* pooled = wpt + (size_t)OUT_ * H_;     // [G,H] bf16

  const int TB = 256;
  const int nodeThr = Nn_ * 128;       // 4 feats per thread
  const int edgeThr = E_ * 128;

  // --- one-shot weight convert+transpose (bf16, [N,K] layout)
  for (int i = 0; i < L_; ++i)
    transpose_bf16<<<(H_ * H_ + TB - 1) / TB, TB, 0, stream>>>(
        W + (size_t)i * H_ * H_, wt + (size_t)i * H_ * H_, H_, H_);
  transpose_bf16<<<(H_ * OUT_ + TB - 1) / TB, TB, 0, stream>>>(Wp, wpt, H_, OUT_);

  // --- degrees & graph segment bounds
  zero_i32<<<(Nn_ + TB - 1) / TB, TB, 0, stream>>>(deg, Nn_);
  deg_count<<<(E_ + TB - 1) / TB, TB, 0, stream>>>(dst, deg);
  deg_recip<<<(Nn_ + TB - 1) / TB, TB, 0, stream>>>(deg, invdeg);
  bounds_init<<<1, 256, 0, stream>>>(gs, ge);
  bounds_scan<<<(Nn_ + TB - 1) / TB, TB, 0, stream>>>(gid, gs, ge);

  // --- atom encoder -> h0
  atom_encode<<<nodeThr / TB, TB, 0, stream>>>(h_node, atom_emb, h0);

  float* hcur = h0;
  float* hnxt = h1;
  for (int i = 0; i < L_; ++i) {
    zero_f32<<<(int)((NH + TB - 1) / TB), TB, 0, stream>>>(neigh, (int)NH);
    const float* bemb_i = bond_emb + (size_t)i * BF_ * BV_ * H_;
    edge_msg<<<edgeThr / TB, TB, 0, stream>>>(hcur, bemb_i, h_edge, src, dst, neigh);
    combine_bf16<<<nodeThr / TB, TB, 0, stream>>>(hcur, invdeg, neigh, rbf);

    dim3 gg(H_ / BN, Nn_ / BM);
    wmma_gemm_bf16<<<gg, TB, 0, stream>>>(rbf, wt + (size_t)i * H_ * H_,
                                          bb + (size_t)i * H_, rbuf, Nn_, H_, H_);

    gn_stats<<<G_, TB, 0, stream>>>(rbuf, gs, ge, gn_ms + (size_t)i * H_, mean, var);
    gn_apply<<<nodeThr / TB, TB, 0, stream>>>(rbuf, hcur, gid, mean, var,
                                              gn_w + (size_t)i * H_,
                                              gn_b + (size_t)i * H_,
                                              gn_ms + (size_t)i * H_,
                                              hnxt, (i != L_ - 1) ? 1 : 0);
    float* t = hcur; hcur = hnxt; hnxt = t;
  }

  // --- pooling + final projection (M=256, N=128, K=512 -> same WMMA tile)
  pool_sum_bf16<<<G_, TB, 0, stream>>>(hcur, gs, ge, pooled);
  dim3 gf(OUT_ / BN, G_ / BM);
  wmma_gemm_bf16<<<gf, TB, 0, stream>>>(pooled, wpt, bp, out, G_, OUT_, H_);

  (void)in_sizes; (void)n_in; (void)out_size; (void)ws_size;
}
